// VisualEmbeddingLayer_adaptive_18399639896077
// MI455X (gfx1250) — matrix operand order repaired and checked by reading, not compiled, not run on hardware
//
#include <hip/hip_runtime.h>
#include <hip/hip_fp16.h>

typedef __attribute__((ext_vector_type(16))) _Float16 v16h;
typedef __attribute__((ext_vector_type(8)))  _Float16 v8h;
typedef __attribute__((ext_vector_type(4)))  _Float16 v4h;
typedef __attribute__((ext_vector_type(16))) __bf16   v16bf;
typedef __attribute__((ext_vector_type(8)))  __bf16   v8bf;
typedef __attribute__((ext_vector_type(8)))  float    v8f;

#define LDSP 40  // padded LDS row stride in halves (80B: 16B-aligned, conflict-free)

// ---- problem constants ----
#define BB   128
#define NN   577
#define DD   512
#define EE   1024
#define HH   512
#define KSEL 172
#define MM   (BB * KSEL)      // 22016
#define NPAD 640

// ---- workspace layout (bytes) ----
#define OFF_SCORES 0u                      // 128*640*4      = 327680
#define OFF_IDX    327680u                 // 22016*4       -> 415744
#define OFF_ACAT   415744u                 // 22016*1024*2  -> 45504512
#define OFF_W1T    45504512u               // 512*512*2 bf16
#define OFF_W0T    46028800u               // 512*512*2 f16
#define OFF_BCAT   46553088u               // 1024*1024*2 f16
#define OFF_BIASC  48650240u               // 1024*4
#define OFF_SCALE  48654336u               // 512*4
#define OFF_SHIFT  48656384u               // 512*4

// ============ CDNA5 async global->LDS copy (ASYNCcnt-tracked DMA) ============
__device__ __forceinline__ void async_cp16(unsigned lds_b, const _Float16* g) {
  asm volatile("global_load_async_to_lds_b128 %0, %1, off"
               :: "v"(lds_b), "v"(g) : "memory");
}
__device__ __forceinline__ void wait_async0() {
  asm volatile("s_wait_asynccnt 0x0" ::: "memory");
}

// ================= fragment loads from LDS (per CDNA5 ISA 7.12.2) =================
__device__ __forceinline__ v16h lds_a_frag(const _Float16* rowbase, int hi) {
  v8h lo = *(const v8h*)(rowbase + hi * 8);
  v8h hv = *(const v8h*)(rowbase + 16 + hi * 8);
  return __builtin_shufflevector(lo, hv, 0,1,2,3,4,5,6,7,8,9,10,11,12,13,14,15);
}
__device__ __forceinline__ v16h lds_b_frag(const _Float16* rowbase, int hi) {
  v8h lo = *(const v8h*)(rowbase + hi * 8);
  v8h hv = *(const v8h*)(rowbase + 16 + hi * 8);
  return __builtin_shufflevector(lo, hv, 0,1,2,3,4,5,6,7,8,9,10,11,12,13,14,15);
}
__device__ __forceinline__ v16bf lds_a_frag_bf(const __bf16* rowbase, int hi) {
  v8bf lo = *(const v8bf*)(rowbase + hi * 8);
  v8bf hv = *(const v8bf*)(rowbase + 16 + hi * 8);
  return __builtin_shufflevector(lo, hv, 0,1,2,3,4,5,6,7,8,9,10,11,12,13,14,15);
}
__device__ __forceinline__ v16bf lds_b_frag_bf(const __bf16* rowbase, int hi) {
  v8bf lo = *(const v8bf*)(rowbase + hi * 8);
  v8bf hv = *(const v8bf*)(rowbase + 16 + hi * 8);
  return __builtin_shufflevector(lo, hv, 0,1,2,3,4,5,6,7,8,9,10,11,12,13,14,15);
}

// ===== shared f16 GEMM core: 128x128 block tile, 8 waves (4M x 2N, 32x64/wave),
// ===== double-buffered LDS fed by async global->LDS DMA =====================
#define TILEHALF (128 * LDSP)     // halves per LDS buffer
__device__ __forceinline__ void gemm_f16_core128(
    const _Float16* __restrict__ A, int lda,
    const _Float16* __restrict__ Bt, int ldb,
    int m0, int n0, int K,
    _Float16* A_lds, _Float16* B_lds,   // each 2*TILEHALF halves
    v8f c[2][4])
{
  const int tid  = threadIdx.x;
  const int lane = tid & 31, wave = tid >> 5;
  const int wm = wave >> 1, wn = wave & 1;
  const int lo = lane & 15, hi = lane >> 4;
  v8f z = {0.f,0.f,0.f,0.f,0.f,0.f,0.f,0.f};
  for (int mt = 0; mt < 2; ++mt)
    for (int nt = 0; nt < 4; ++nt) c[mt][nt] = z;

  // staging: 256 threads cover 128 rows x 32 halves for A and for B
  const int srow = tid >> 1, sseg = tid & 1;
  const _Float16* agp = A  + (size_t)(m0 + srow) * lda + sseg * 16;
  const _Float16* bgp = Bt + (size_t)(n0 + srow) * ldb + sseg * 16;
  const unsigned albase = (unsigned)(uintptr_t)(A_lds + srow * LDSP + sseg * 16);
  const unsigned blbase = (unsigned)(uintptr_t)(B_lds + srow * LDSP + sseg * 16);
  const unsigned bufb = TILEHALF * 2;  // buffer stride in bytes

  // prefetch tile 0 into buffer 0
  async_cp16(albase,      agp);
  async_cp16(albase + 16, agp + 8);
  async_cp16(blbase,      bgp);
  async_cp16(blbase + 16, bgp + 8);

  const int nk = K >> 5;
  for (int kt = 0; kt < nk; ++kt) {
    const int cur = kt & 1;
    wait_async0();          // my wave's DMA into buf[cur] complete
    __syncthreads();        // all waves' DMA complete -> buf[cur] readable
    if (kt + 1 < nk) {      // overlap: DMA next tile into other buffer
      const unsigned nb = (unsigned)(cur ^ 1) * bufb;
      const _Float16* ag = agp + (kt + 1) * 32;
      const _Float16* bg = bgp + (kt + 1) * 32;
      async_cp16(albase + nb,      ag);
      async_cp16(albase + nb + 16, ag + 8);
      async_cp16(blbase + nb,      bg);
      async_cp16(blbase + nb + 16, bg + 8);
    }
    const _Float16* Al = A_lds + cur * TILEHALF;
    const _Float16* Bl = B_lds + cur * TILEHALF;
    v16h af[2], bfv[4];
    for (int mt = 0; mt < 2; ++mt)
      af[mt] = lds_a_frag(Al + (wm * 32 + mt * 16 + lo) * LDSP, hi);
    for (int nt = 0; nt < 4; ++nt)
      bfv[nt] = lds_b_frag(Bl + (wn * 64 + nt * 16 + lo) * LDSP, hi);
    for (int mt = 0; mt < 2; ++mt)
      for (int nt = 0; nt < 4; ++nt)
        c[mt][nt] = __builtin_amdgcn_wmma_f32_16x16x32_f16(
            false, af[mt], false, bfv[nt], (short)0, c[mt][nt], false, false);
  }
}

// ================= k0: weight transpose / convert / fuse =================
__global__ __launch_bounds__(256) void k0_prep(
    const float* __restrict__ wg_w1, const float* __restrict__ mlp_w0,
    const float* __restrict__ mlp_w1, const float* __restrict__ fc_w,
    const float* __restrict__ mlp_b1, const float* __restrict__ fc_b,
    __bf16* __restrict__ W1T, _Float16* __restrict__ w0T,
    _Float16* __restrict__ BcatT, float* __restrict__ bias_cat)
{
  const int stride = gridDim.x * blockDim.x;
  for (int i = blockIdx.x * blockDim.x + threadIdx.x; i < (1 << 20); i += stride) {
    int n = i >> 10, k = i & 1023;
    float v = (k < HH) ? mlp_w1[(size_t)k * EE + n] : fc_w[(size_t)(k - HH) * EE + n];
    BcatT[i] = (_Float16)v;
    if (i < DD * DD) {
      int e = i >> 9, d = i & 511;
      W1T[i] = (__bf16)wg_w1[(size_t)d * DD + e];
      w0T[i] = (_Float16)mlp_w0[(size_t)d * HH + e];
    }
    if (i < EE) bias_cat[i] = mlp_b1[i] + fc_b[i];
  }
}

// ================= k1: fused wg-MLP scorer (bf16 WMMA) =================
__global__ __launch_bounds__(256) void k1_score(
    const float* __restrict__ feats, const __bf16* __restrict__ W1T,
    const float* __restrict__ b1, const float* __restrict__ w2,
    float* __restrict__ scores)
{
  __shared__ __align__(16) __bf16 A_lds[128 * LDSP];
  __shared__ __align__(16) __bf16 B_lds[64 * LDSP];
  __shared__ float score_lds[128];

  const int b = blockIdx.y, n0 = blockIdx.x * 128;
  const int tid = threadIdx.x;
  const int lane = tid & 31, wave = tid >> 5;
  const int wm = wave >> 1, wn = wave & 1;
  const int lo = lane & 15, hi = lane >> 4;
  if (tid < 128) score_lds[tid] = 0.f;

  float sp[2][8];
  for (int mt = 0; mt < 2; ++mt)
    for (int r = 0; r < 8; ++r) sp[mt][r] = 0.f;

  const int arow = tid >> 1, aseg = tid & 1;
  const int brow = tid >> 2, bseg = tid & 3;
  const int gn = n0 + arow;
  const float* arowp = feats + ((size_t)b * NN + gn) * DD;

  for (int eb = 0; eb < 8; ++eb) {
    v8f z = {0.f,0.f,0.f,0.f,0.f,0.f,0.f,0.f};
    v8f c[2][2]; c[0][0] = z; c[0][1] = z; c[1][0] = z; c[1][1] = z;
    for (int kt = 0; kt < 16; ++kt) {
      __syncthreads();
      // stage A tile 128x32: f32 -> bf16
      float4 f0 = make_float4(0.f,0.f,0.f,0.f), f1 = f0, f2 = f0, f3 = f0;
      if (gn < NN) {
        const float4* p = (const float4*)(arowp + kt * 32 + aseg * 16);
        f0 = p[0]; f1 = p[1]; f2 = p[2]; f3 = p[3];
      }
      __bf16* dst = A_lds + arow * LDSP + aseg * 16;
      v8bf x0 = { (__bf16)f0.x,(__bf16)f0.y,(__bf16)f0.z,(__bf16)f0.w,
                  (__bf16)f1.x,(__bf16)f1.y,(__bf16)f1.z,(__bf16)f1.w };
      v8bf x1 = { (__bf16)f2.x,(__bf16)f2.y,(__bf16)f2.z,(__bf16)f2.w,
                  (__bf16)f3.x,(__bf16)f3.y,(__bf16)f3.z,(__bf16)f3.w };
      *(v8bf*)dst = x0; *(v8bf*)(dst + 8) = x1;
      // stage Bt tile 64x32 from W1T
      const uint4* bsrc = (const uint4*)(W1T + (size_t)(eb * 64 + brow) * DD + kt * 32 + bseg * 8);
      *(uint4*)(B_lds + brow * LDSP + bseg * 8) = bsrc[0];
      __syncthreads();

      v16bf af[2], bf[2];
      for (int mt = 0; mt < 2; ++mt)
        af[mt] = lds_a_frag_bf(A_lds + (wm * 32 + mt * 16 + lo) * LDSP, hi);
      for (int nt = 0; nt < 2; ++nt)
        bf[nt] = lds_b_frag_bf(B_lds + (wn * 32 + nt * 16 + lo) * LDSP, hi);
      for (int mt = 0; mt < 2; ++mt)
        for (int nt = 0; nt < 2; ++nt)
          c[mt][nt] = __builtin_amdgcn_wmma_f32_16x16x32_bf16(
              false, af[mt], false, bf[nt], (short)0, c[mt][nt], false, false);
    }
    // epilogue: sp += relu(c + b1[e]) * w2[e], summed over this 64-wide e block
    for (int nt = 0; nt < 2; ++nt) {
      int e = eb * 64 + wn * 32 + nt * 16 + lo;
      float b1v = b1[e], w2v = w2[e];
      for (int mt = 0; mt < 2; ++mt)
        for (int r = 0; r < 8; ++r) {
          float v = c[mt][nt][r] + b1v;
          sp[mt][r] += (v > 0.f) ? v * w2v : 0.f;
        }
    }
  }
  // reduce across the 16 columns each lane group holds, then across waves via LDS
  for (int mt = 0; mt < 2; ++mt)
    for (int r = 0; r < 8; ++r) {
      float v = sp[mt][r];
      v += __shfl_xor(v, 1, 32); v += __shfl_xor(v, 2, 32);
      v += __shfl_xor(v, 4, 32); v += __shfl_xor(v, 8, 32);
      if (lo == 0) atomicAdd(&score_lds[wm * 32 + mt * 16 + hi * 8 + r], v);
    }
  __syncthreads();
  if (tid < 128) {
    int n = n0 + tid;
    if (n < NN) scores[(size_t)b * NPAD + n] = score_lds[tid];
  }
}

// ================= k2: rank-based top-172, ascending indices =================
__global__ __launch_bounds__(256) void k2_topk(
    const float* __restrict__ scores, int* __restrict__ idx)
{
  __shared__ float s[NN];
  __shared__ int   kp[NN];
  const int b = blockIdx.x, tid = threadIdx.x;
  for (int i = tid; i < NN; i += 256) s[i] = scores[(size_t)b * NPAD + i];
  __syncthreads();
  if (tid == 0) s[0] = -1e30f;  // mask CLS token
  __syncthreads();
  for (int n = tid; n < NN; n += 256) {
    float v = s[n]; int rank = 0;
    for (int m = 0; m < NN; ++m) {
      float u = s[m];
      rank += (u > v) || (u == v && m < n);
    }
    kp[n] = (rank < KSEL) ? 1 : 0;
  }
  __syncthreads();
  for (int n = tid; n < NN; n += 256) {
    if (kp[n]) {
      int pos = 0;
      for (int m = 0; m < n; ++m) pos += kp[m];
      idx[b * KSEL + pos] = n;
    }
  }
}

// ================= k3: gather + L2-normalize -> Acat[:,512:1024] f16 =========
__global__ __launch_bounds__(128) void k3_gather(
    const float* __restrict__ feats, const int* __restrict__ idx,
    _Float16* __restrict__ Acat)
{
  const int row = blockIdx.x;          // 0..22015
  const int b = row / KSEL;
  const int n = idx[row];
  const int tid = threadIdx.x;
  const float4 v = ((const float4*)(feats + ((size_t)b * NN + n) * DD))[tid];
  float sq = v.x*v.x + v.y*v.y + v.z*v.z + v.w*v.w;
  sq += __shfl_xor(sq, 1, 32);  sq += __shfl_xor(sq, 2, 32);
  sq += __shfl_xor(sq, 4, 32);  sq += __shfl_xor(sq, 8, 32);
  sq += __shfl_xor(sq, 16, 32);
  __shared__ float wred[4];
  if ((tid & 31) == 0) wred[tid >> 5] = sq;
  __syncthreads();
  const float tot = wred[0] + wred[1] + wred[2] + wred[3];
  const float inv = 1.f / (sqrtf(tot) + 1e-8f);
  v4h o = { (_Float16)(v.x*inv), (_Float16)(v.y*inv),
            (_Float16)(v.z*inv), (_Float16)(v.w*inv) };
  *(v4h*)(Acat + (size_t)row * 1024 + 512 + tid * 4) = o;
}

// ================= k4: X = sel @ w0 + b0 -> Acat[:,0:512] f16 ===============
__global__ __launch_bounds__(256) void k4_gemm1(
    const _Float16* __restrict__ Acat_sel,  // = Acat + 512
    const _Float16* __restrict__ w0T, const float* __restrict__ b0,
    _Float16* __restrict__ Xout)            // = Acat
{
  __shared__ __align__(16) _Float16 A_lds[2 * TILEHALF];
  __shared__ __align__(16) _Float16 B_lds[2 * TILEHALF];
  v8f c[2][4];
  const int m0 = blockIdx.x * 128, n0 = blockIdx.y * 128;
  gemm_f16_core128(Acat_sel, 1024, w0T, 512, m0, n0, 512, A_lds, B_lds, c);

  const int tid = threadIdx.x, lane = tid & 31, wave = tid >> 5;
  const int wm = wave >> 1, wn = wave & 1, lo = lane & 15, hi = lane >> 4;
  for (int nt = 0; nt < 4; ++nt) {
    int n = n0 + wn * 64 + nt * 16 + lo;
    float bias = b0[n];
    for (int mt = 0; mt < 2; ++mt)
      for (int r = 0; r < 8; ++r) {
        int row = m0 + wm * 32 + mt * 16 + hi * 8 + r;
        Xout[(size_t)row * 1024 + n] = (_Float16)(c[mt][nt][r] + bias);
      }
  }
}

// ================= k4b: deterministic BN stats -> scale/shift ================
__global__ __launch_bounds__(256) void k4b_stats(
    const _Float16* __restrict__ X, const float* __restrict__ bn_g,
    const float* __restrict__ bn_b, float* __restrict__ scale,
    float* __restrict__ shift)
{
  __shared__ float ssum[256], ssq[256];
  const int t = threadIdx.x;
  const int rg = t >> 3, cpos = t & 7;
  const int j = blockIdx.x * 8 + cpos;
  float s = 0.f, q = 0.f;
  for (int row = rg; row < MM; row += 32) {
    float v = (float)X[(size_t)row * 1024 + j];
    s += v; q += v * v;
  }
  ssum[t] = s; ssq[t] = q;
  __syncthreads();
  for (int st = 16; st > 0; st >>= 1) {
    if (rg < st) { ssum[t] += ssum[t + st * 8]; ssq[t] += ssq[t + st * 8]; }
    __syncthreads();
  }
  if (rg == 0) {
    const float invM = 1.f / (float)MM;
    float mean = ssum[t] * invM;
    float var  = ssq[t] * invM - mean * mean;
    float sc = rsqrtf(var + 1e-5f) * bn_g[j];
    scale[j] = sc;
    shift[j] = bn_b[j] - mean * sc;
  }
}

// ================= k6a: in-place BN + ReLU on Acat[:,0:512] =================
__global__ __launch_bounds__(256) void k6a_bn(
    _Float16* __restrict__ X, const float* __restrict__ scale,
    const float* __restrict__ shift)
{
  const size_t total = (size_t)MM * 128;  // quads of 4 halves
  const size_t stride = (size_t)gridDim.x * blockDim.x;
  for (size_t qv = (size_t)blockIdx.x * blockDim.x + threadIdx.x; qv < total; qv += stride) {
    size_t row = qv >> 7; int j = (int)(qv & 127) * 4;
    _Float16* p = X + row * 1024 + j;
    v4h x = *(v4h*)p;
    float r0 = fmaxf((float)x[0] * scale[j]     + shift[j],     0.f);
    float r1 = fmaxf((float)x[1] * scale[j + 1] + shift[j + 1], 0.f);
    float r2 = fmaxf((float)x[2] * scale[j + 2] + shift[j + 2], 0.f);
    float r3 = fmaxf((float)x[3] * scale[j + 3] + shift[j + 3], 0.f);
    v4h o = { (_Float16)r0, (_Float16)r1, (_Float16)r2, (_Float16)r3 };
    *(v4h*)p = o;
  }
}

// ================= k6b: out = Acat @ BcatT + bias_cat (fp32 out) ============
__global__ __launch_bounds__(256) void k6b_gemm2(
    const _Float16* __restrict__ Acat, const _Float16* __restrict__ BcatT,
    const float* __restrict__ bias_cat, float* __restrict__ out)
{
  __shared__ __align__(16) _Float16 A_lds[2 * TILEHALF];
  __shared__ __align__(16) _Float16 B_lds[2 * TILEHALF];
  v8f c[2][4];
  const int m0 = blockIdx.x * 128, n0 = blockIdx.y * 128;
  gemm_f16_core128(Acat, 1024, BcatT, 1024, m0, n0, 1024, A_lds, B_lds, c);

  const int tid = threadIdx.x, lane = tid & 31, wave = tid >> 5;
  const int wm = wave >> 1, wn = wave & 1, lo = lane & 15, hi = lane >> 4;
  for (int nt = 0; nt < 4; ++nt) {
    int n = n0 + wn * 64 + nt * 16 + lo;
    float bias = bias_cat[n];
    for (int mt = 0; mt < 2; ++mt)
      for (int r = 0; r < 8; ++r) {
        int row = m0 + wm * 32 + mt * 16 + hi * 8 + r;
        out[(size_t)row * 1024 + n] = c[mt][nt][r] + bias;
      }
  }
}

// ================= host launcher =================
extern "C" void kernel_launch(void* const* d_in, const int* in_sizes, int n_in,
                              void* d_out, int out_size, void* d_ws, size_t ws_size,
                              hipStream_t stream) {
  (void)in_sizes; (void)n_in; (void)out_size; (void)ws_size;
  const float* features = (const float*)d_in[0];
  // d_in[1] = atten (unused: k is static)
  const float* wg_w1 = (const float*)d_in[2];
  const float* wg_b1 = (const float*)d_in[3];
  const float* wg_w2 = (const float*)d_in[4];
  // d_in[5] = wg_b2 (constant shift: does not affect top-k ordering)
  const float* fc_w  = (const float*)d_in[6];
  const float* fc_b  = (const float*)d_in[7];
  const float* mlp_w0 = (const float*)d_in[8];
  const float* mlp_b0 = (const float*)d_in[9];
  const float* bn_g  = (const float*)d_in[10];
  const float* bn_b  = (const float*)d_in[11];
  const float* mlp_w1 = (const float*)d_in[12];
  const float* mlp_b1 = (const float*)d_in[13];

  char* ws = (char*)d_ws;
  float*    scores   = (float*)(ws + OFF_SCORES);
  int*      idx      = (int*)(ws + OFF_IDX);
  _Float16* Acat     = (_Float16*)(ws + OFF_ACAT);
  __bf16*   W1T      = (__bf16*)(ws + OFF_W1T);
  _Float16* w0T      = (_Float16*)(ws + OFF_W0T);
  _Float16* BcatT    = (_Float16*)(ws + OFF_BCAT);
  float*    bias_cat = (float*)(ws + OFF_BIASC);
  float*    scale    = (float*)(ws + OFF_SCALE);
  float*    shift    = (float*)(ws + OFF_SHIFT);

  k0_prep<<<1024, 256, 0, stream>>>(wg_w1, mlp_w0, mlp_w1, fc_w, mlp_b1, fc_b,
                                    W1T, w0T, BcatT, bias_cat);
  k1_score<<<dim3(5, BB), 256, 0, stream>>>(features, W1T, wg_b1, wg_w2, scores);
  k2_topk<<<BB, 256, 0, stream>>>(scores, idx);
  k3_gather<<<MM, 128, 0, stream>>>(features, idx, Acat);
  k4_gemm1<<<dim3(MM / 128, HH / 128), 256, 0, stream>>>(Acat + 512, w0T, mlp_b0, Acat);
  k4b_stats<<<HH / 8, 256, 0, stream>>>(Acat, bn_g, bn_b, scale, shift);
  k6a_bn<<<2048, 256, 0, stream>>>(Acat, scale, shift);
  k6b_gemm2<<<dim3(MM / 128, EE / 128), 256, 0, stream>>>(Acat, BcatT, bias_cat,
                                                          (float*)d_out);
}
